// ViTBlock_23072564314559
// MI455X (gfx1250) — compile-verified, matrix operand order checked by reading, not run on hardware
//
#include <hip/hip_runtime.h>
#include <math.h>

// ---------------------------------------------------------------------------
// Types for CDNA5 WMMA (wave32): v_wmma_f32_16x16x32_bf16
// ---------------------------------------------------------------------------
typedef __attribute__((ext_vector_type(16))) __bf16 v16bf;
typedef __attribute__((ext_vector_type(8)))  __bf16 v8bf;
typedef __attribute__((ext_vector_type(8)))  float  v8f;

#define WMMA_BF16(a, b, c) \
  __builtin_amdgcn_wmma_f32_16x16x32_bf16(false, (a), false, (b), (short)0, (c), false, false)

// CDNA5 async global->LDS copy (ASYNCcnt path), with portable fallback.
#if defined(__has_builtin)
#if __has_builtin(__builtin_amdgcn_global_load_async_to_lds_b128) && \
    __has_builtin(__builtin_amdgcn_s_wait_asynccnt)
#define HAVE_ASYNC 1
#endif
#endif

#ifdef HAVE_ASYNC
typedef int v4i __attribute__((vector_size(16)));
typedef __attribute__((address_space(1))) v4i* gv4i_p;   // global
typedef __attribute__((address_space(3))) v4i* lv4i_p;   // LDS
#endif

__device__ inline v8f v8f_zero() {
  v8f z = {0.f, 0.f, 0.f, 0.f, 0.f, 0.f, 0.f, 0.f};
  return z;
}

// Load a 16x32 bf16 A-fragment (or B-fragment from an N-major/K-contiguous
// matrix) per ISA 7.12.2: lane = half*16 + idx; row = idx; K chunks at
// {half*8 .. half*8+7} and {16+half*8 .. 16+half*8+7}; each chunk is a
// contiguous 16-byte load.
__device__ inline v16bf load_frag(const __bf16* p, int ld, int lane) {
  int idx = lane & 15;
  int hf  = lane >> 4;
  const __bf16* base = p + (size_t)idx * ld + hf * 8;
  v8bf lo = *(const v8bf*)(base);
  v8bf hi = *(const v8bf*)(base + 16);
  return __builtin_shufflevector(lo, hi, 0, 1, 2, 3, 4, 5, 6, 7,
                                 8, 9, 10, 11, 12, 13, 14, 15);
}

// ---------------------------------------------------------------------------
// Problem constants
// ---------------------------------------------------------------------------
#define BATCH 16
#define IMGS  56
#define DIM   768
#define NHEAD 12
#define HDIM  64
#define WSZ   14
#define MLPD  3072
#define TOK   196          // tokens per window
#define TOKP  208          // padded tokens per window (13*16)
#define NWIN  256          // total windows
#define NWH   3072         // windows * heads
#define MROWS (NWIN * TOKP)          // 53248 padded rows (window layout)
#define NTOKF (BATCH * IMGS * IMGS)  // 50176 flat rows

// ---------------------------------------------------------------------------
// fp32 (K x N) -> bf16 transposed (N x K)
// ---------------------------------------------------------------------------
__global__ void convT_kernel(const float* __restrict__ in, __bf16* __restrict__ out,
                             int K, int N) {
  size_t total = (size_t)K * N;
  for (size_t e = (size_t)blockIdx.x * blockDim.x + threadIdx.x; e < total;
       e += (size_t)gridDim.x * blockDim.x) {
    size_t n = e / K, k = e - n * K;
    out[e] = (__bf16)in[k * N + n];
  }
}

// ---------------------------------------------------------------------------
// LayerNorm over 768 dims; one block (256 thr) per row; optional window remap
// ---------------------------------------------------------------------------
__device__ inline float block_sum768(float v, float* sh) {
#pragma unroll
  for (int d = 16; d > 0; d >>= 1) v += __shfl_xor(v, d, 32);
  int w = threadIdx.x >> 5;
  if ((threadIdx.x & 31) == 0) sh[w] = v;
  __syncthreads();
  if (threadIdx.x == 0) {
    float t = 0.f;
    for (int i = 0; i < 8; ++i) t += sh[i];
    sh[8] = t;
  }
  __syncthreads();
  float r = sh[8];
  __syncthreads();
  return r;
}

__global__ __launch_bounds__(256) void ln_kernel(const float* __restrict__ x,
                                                 const float* __restrict__ gw,
                                                 const float* __restrict__ bw,
                                                 __bf16* __restrict__ out,
                                                 int winmap) {
  __shared__ float sh[9];
  int row = blockIdx.x;
  const float* rp = x + (size_t)row * DIM;
  int t = threadIdx.x;
  float a0 = rp[t], a1 = rp[t + 256], a2 = rp[t + 512];
  float mu = block_sum768(a0 + a1 + a2, sh) * (1.0f / 768.0f);
  float d0 = a0 - mu, d1 = a1 - mu, d2 = a2 - mu;
  float var = block_sum768(d0 * d0 + d1 * d1 + d2 * d2, sh) * (1.0f / 768.0f);
  float rstd = rsqrtf(var + 1e-5f);
  size_t orow;
  if (winmap) {
    int bimg = row / (IMGS * IMGS), rem = row - bimg * IMGS * IMGS;
    int ih = rem / IMGS, iw = rem - ih * IMGS;
    int wh = ih / WSZ, th = ih - wh * WSZ;
    int ww = iw / WSZ, tw = iw - ww * WSZ;
    int window = (bimg * 4 + wh) * 4 + ww;
    orow = (size_t)window * TOKP + th * WSZ + tw;
  } else {
    orow = (size_t)row;
  }
  __bf16* op = out + orow * DIM;
  op[t]       = (__bf16)(d0 * rstd * gw[t]       + bw[t]);
  op[t + 256] = (__bf16)(d1 * rstd * gw[t + 256] + bw[t + 256]);
  op[t + 512] = (__bf16)(d2 * rstd * gw[t + 512] + bw[t + 512]);
}

// Zero the 12 padding rows of every window in xw (window layout)
__global__ void padzero_kernel(__bf16* __restrict__ xw) {
  const int total = NWIN * (TOKP - TOK) * DIM;
  int e = blockIdx.x * blockDim.x + threadIdx.x;
  if (e < total) {
    int window = e / ((TOKP - TOK) * DIM);
    int r = e - window * (TOKP - TOK) * DIM;
    int token = TOK + r / DIM;
    int c = r - (r / DIM) * DIM;
    xw[((size_t)window * TOKP + token) * DIM + c] = (__bf16)0.f;
  }
}

// ---------------------------------------------------------------------------
// Generic WMMA GEMM: C(MxN) = A(MxK,bf16,row-major) * Bt(NxK,bf16)^T
// Block: 256 threads = 8 waves (4 row x 2 col); block tile 128(M) x 128(N);
// wave tile 32x64. B panel (128x32 bf16 = 8 KB) is double-buffered in LDS,
// filled with async global->LDS copies (ASYNCcnt) when available.
// ---------------------------------------------------------------------------
enum { EPI_QKV = 0, EPI_PROJ = 1, EPI_L1 = 2, EPI_L2 = 3 };

struct GArgs {
  const __bf16* A;
  const __bf16* Bt;
  int K;
  const float* bias;
  const float* resid;   // PROJ: x ; L2: hidden
  float* outf;          // PROJ: hidden ; L2: d_out
  __bf16* outb;         // L1: mbuf
  __bf16* q;            // QKV outputs
  __bf16* k;
  __bf16* v;
};

// Stage one 128x32 B panel (N-major, ld=32) into LDS. 512 16-byte chunks,
// 2 per thread. Chunk c -> n_local = c>>2, koff = (c&3)*8.
__device__ inline void stage_b(__bf16* dst, const __bf16* Bt, size_t nblock,
                               int k0, int K, int tid) {
#pragma unroll
  for (int h = 0; h < 2; ++h) {
    int c = tid + h * 256;
    int nl = c >> 2;
    int ko = (c & 3) * 8;
    const __bf16* src = Bt + (nblock + nl) * (size_t)K + k0 + ko;
    __bf16* d = dst + nl * 32 + ko;
#ifdef HAVE_ASYNC
    __builtin_amdgcn_global_load_async_to_lds_b128((gv4i_p)(src), (lv4i_p)(d), 0, 0);
#else
    *(v8bf*)d = *(const v8bf*)src;
#endif
  }
}

template <int EPI>
__global__ __launch_bounds__(256) void gemm_kernel(GArgs g) {
  __shared__ __bf16 btile[2][128 * 32];   // 2 x 8 KB

  int tid = threadIdx.x;
  int wid = tid >> 5;
  int lane = tid & 31;
  int idx = lane & 15, hf = lane >> 4;
  int m0 = blockIdx.y * 128 + (wid & 3) * 32;
  int cg = wid >> 2;                       // 0..1
  int n0 = blockIdx.x * 128 + cg * 64;
  const int K = g.K;
  size_t nblock = (size_t)blockIdx.x * 128;

  v8f c[2][4];
#pragma unroll
  for (int i = 0; i < 2; ++i)
#pragma unroll
    for (int j = 0; j < 4; ++j) c[i][j] = v8f_zero();

  const __bf16* A0 = g.A + (size_t)m0 * K;
  const __bf16* A1 = g.A + (size_t)(m0 + 16) * K;

  int nsteps = K / 32;
  stage_b(&btile[0][0], g.Bt, nblock, 0, K, tid);

  for (int s = 0; s < nsteps; ++s) {
    int k0 = s * 32;
    int cur = s & 1;
    if (s + 1 < nsteps) {
      stage_b(&btile[cur ^ 1][0], g.Bt, nblock, k0 + 32, K, tid);
#ifdef HAVE_ASYNC
      __builtin_amdgcn_s_wait_asynccnt(2);  // drain current panel's 2 copies
#endif
    } else {
#ifdef HAVE_ASYNC
      __builtin_amdgcn_s_wait_asynccnt(0);
#endif
    }
    __syncthreads();

    v16bf a0 = load_frag(A0 + k0, K, lane);
    v16bf a1 = load_frag(A1 + k0, K, lane);
    const __bf16* bp = &btile[cur][cg * 64 * 32];
#pragma unroll
    for (int j = 0; j < 4; ++j) {
      v16bf b = load_frag(bp + j * 16 * 32, 32, lane);
      c[0][j] = WMMA_BF16(a0, b, c[0][j]);
      c[1][j] = WMMA_BF16(a1, b, c[1][j]);
    }
    __syncthreads();  // all waves done reading before panel is overwritten
  }

#pragma unroll
  for (int i = 0; i < 2; ++i) {
#pragma unroll
    for (int j = 0; j < 4; ++j) {
#pragma unroll
      for (int v = 0; v < 8; ++v) {
        int m = m0 + i * 16 + hf * 8 + v;
        int n = n0 + j * 16 + idx;
        float val = c[i][j][v] + g.bias[n];
        if (EPI == EPI_QKV) {
          int window = m / TOKP, token = m - window * TOKP;
          int which = n / DIM, rem = n - which * DIM;
          int head = rem >> 6, hd = rem & 63;
          size_t wh = (size_t)window * NHEAD + head;
          if (which == 0)
            g.q[(wh * TOKP + token) * HDIM + hd] = (__bf16)val;
          else if (which == 1)
            g.k[(wh * TOKP + token) * HDIM + hd] = (__bf16)val;
          else
            g.v[(wh * HDIM + hd) * TOKP + token] = (__bf16)val;
        } else if (EPI == EPI_PROJ) {
          int window = m / TOKP, token = m - window * TOKP;
          if (token < TOK) {
            int bimg = window >> 4, wrem = window & 15;
            int wh = wrem >> 2, ww = wrem & 3;
            int th = token / WSZ, tw = token - th * WSZ;
            size_t fl = (((size_t)bimg * IMGS + (wh * WSZ + th)) * IMGS +
                         (ww * WSZ + tw)) * DIM + n;
            g.outf[fl] = val + g.resid[fl];
          }
        } else if (EPI == EPI_L1) {
          float ge = 0.5f * val * (1.0f + erff(val * 0.70710678118f));
          g.outb[(size_t)m * MLPD + n] = (__bf16)ge;
        } else {  // EPI_L2
          size_t fl = (size_t)m * DIM + n;
          g.outf[fl] = val + g.resid[fl];
        }
      }
    }
  }
}

// ---------------------------------------------------------------------------
// Attention: one block per (window, head). 13 waves, one 16-query tile each.
// Q,K: (TOKP x 64) token-major bf16 (unscaled). Vt: (64 x TOKP) hd-major.
// S = (Q*scale) Kt + rel bias ; softmax ; O = P V ; write bf16 to ao.
// ---------------------------------------------------------------------------
__global__ __launch_bounds__(416) void attn_kernel(const __bf16* __restrict__ Q,
                                                   const __bf16* __restrict__ Kb,
                                                   const __bf16* __restrict__ Vt,
                                                   const float* __restrict__ relh,
                                                   const float* __restrict__ relw,
                                                   __bf16* __restrict__ O) {
  __shared__ float rel_s[2 * TOK * WSZ];        // rel_h then rel_w, f32
  __shared__ __bf16 pbuf[13 * 16 * 32];         // per-wave P staging slots

  int whd = blockIdx.x;
  int head = whd % NHEAD;
  int win = whd / NHEAD;
  const __bf16* qbase = Q + (size_t)whd * TOKP * HDIM;
  const __bf16* kbase = Kb + (size_t)whd * TOKP * HDIM;
  const __bf16* vbase = Vt + (size_t)whd * HDIM * TOKP;

  // decomposed rel-pos bias tables: rel_s[t*14+kh] = q[t,:] . table[qh-kh+13,:]
  for (int e = threadIdx.x; e < 2 * TOK * WSZ; e += 416) {
    int hsel = e / (TOK * WSZ);
    int r = e - hsel * TOK * WSZ;
    int t = r / WSZ, kj = r - (r / WSZ) * WSZ;
    int qc = hsel ? (t % WSZ) : (t / WSZ);
    const float* tab = (hsel ? relw : relh) + (size_t)(qc - kj + WSZ - 1) * HDIM;
    const __bf16* qrow = qbase + (size_t)t * HDIM;
    float s = 0.f;
#pragma unroll 8
    for (int cc = 0; cc < HDIM; ++cc) s += (float)qrow[cc] * tab[cc];
    rel_s[e] = s;
  }
  __syncthreads();

  int wid = threadIdx.x >> 5;
  int lane = threadIdx.x & 31;
  int idx = lane & 15, hf = lane >> 4;
  int m0 = wid * 16;

  v16bf aq0 = load_frag(qbase + (size_t)m0 * HDIM, HDIM, lane);
  v16bf aq1 = load_frag(qbase + (size_t)m0 * HDIM + 32, HDIM, lane);

  v8f sacc[13];
#pragma unroll
  for (int nt = 0; nt < 13; ++nt) sacc[nt] = v8f_zero();

#pragma unroll
  for (int nt = 0; nt < 13; ++nt) {
    v16bf bk0 = load_frag(kbase + (size_t)(nt * 16) * HDIM, HDIM, lane);
    v16bf bk1 = load_frag(kbase + (size_t)(nt * 16) * HDIM + 32, HDIM, lane);
    sacc[nt] = WMMA_BF16(aq0, bk0, sacc[nt]);
    sacc[nt] = WMMA_BF16(aq1, bk1, sacc[nt]);
  }

  // softmax over 208 (196 valid) keys; rows live across 16-lane halves
  const float scale = 0.125f;  // 64^-0.5
  float rcp[8];
#pragma unroll
  for (int v = 0; v < 8; ++v) {
    int m = m0 + hf * 8 + v;
    bool mrow = (m < TOK);
    float mx = -3.0e30f;
#pragma unroll
    for (int nt = 0; nt < 13; ++nt) {
      int tk = nt * 16 + idx;
      float bias;
      if (tk >= TOK)
        bias = -1.0e30f;
      else if (mrow)
        bias = rel_s[m * WSZ + tk / WSZ] +
               rel_s[TOK * WSZ + m * WSZ + (tk - (tk / WSZ) * WSZ)];
      else
        bias = 0.f;
      float val = sacc[nt][v] * scale + bias;
      sacc[nt][v] = val;
      mx = fmaxf(mx, val);
    }
#pragma unroll
    for (int d = 1; d < 16; d <<= 1) mx = fmaxf(mx, __shfl_xor(mx, d, 32));
    float sum = 0.f;
#pragma unroll
    for (int nt = 0; nt < 13; ++nt) {
      float p = __expf(sacc[nt][v] - mx);
      sacc[nt][v] = p;
      sum += p;
    }
#pragma unroll
    for (int d = 1; d < 16; d <<= 1) sum += __shfl_xor(sum, d, 32);
    rcp[v] = 1.0f / sum;
  }

  // O = P @ V : restage P (C-layout f32) as bf16 A-fragments via private LDS
  v8f oacc[4];
#pragma unroll
  for (int j = 0; j < 4; ++j) oacc[j] = v8f_zero();
  __bf16* slot = &pbuf[wid * 16 * 32];

#pragma unroll
  for (int pr = 0; pr < 7; ++pr) {
#pragma unroll
    for (int t2 = 0; t2 < 2; ++t2) {
      int nt = pr * 2 + t2;
#pragma unroll
      for (int v = 0; v < 8; ++v) {
        float pv = (nt < 13) ? sacc[nt][v] : 0.f;
        slot[(hf * 8 + v) * 32 + t2 * 16 + idx] = (__bf16)pv;
      }
    }
    v16bf ap = load_frag(slot, 32, lane);  // ds_load path, wave-private
#pragma unroll
    for (int j = 0; j < 4; ++j) {
      v16bf bv = load_frag(vbase + (size_t)(j * 16) * TOKP + pr * 32, TOKP, lane);
      oacc[j] = WMMA_BF16(ap, bv, oacc[j]);
    }
  }

#pragma unroll
  for (int j = 0; j < 4; ++j) {
#pragma unroll
    for (int v = 0; v < 8; ++v) {
      int m = m0 + hf * 8 + v;
      if (m < TOK) {
        int col = head * HDIM + j * 16 + idx;
        O[((size_t)win * TOKP + m) * DIM + col] = (__bf16)(oacc[j][v] * rcp[v]);
      }
    }
  }
}

// ---------------------------------------------------------------------------
// Host launcher
// ---------------------------------------------------------------------------
extern "C" void kernel_launch(void* const* d_in, const int* in_sizes, int n_in,
                              void* d_out, int out_size, void* d_ws, size_t ws_size,
                              hipStream_t stream) {
  const float* x     = (const float*)d_in[0];
  const float* n1g   = (const float*)d_in[1];
  const float* n1b   = (const float*)d_in[2];
  const float* qkvw  = (const float*)d_in[3];
  const float* qkvb  = (const float*)d_in[4];
  const float* projw = (const float*)d_in[5];
  const float* projb = (const float*)d_in[6];
  const float* relh  = (const float*)d_in[7];
  const float* relw  = (const float*)d_in[8];
  const float* n2g   = (const float*)d_in[9];
  const float* n2b   = (const float*)d_in[10];
  const float* l1w   = (const float*)d_in[11];
  const float* l1b   = (const float*)d_in[12];
  const float* l2w   = (const float*)d_in[13];
  const float* l2b   = (const float*)d_in[14];
  float* out = (float*)d_out;

  char* ws = (char*)d_ws;
  size_t off = 0;
  auto take = [&](size_t bytes) -> char* {
    char* p = ws + off;
    off += (bytes + 255) & ~(size_t)255;
    return p;
  };

  __bf16* xw   = (__bf16*)take((size_t)MROWS * DIM * 2);           // LN1, windowed
  __bf16* qb   = (__bf16*)take((size_t)NWH * TOKP * HDIM * 2);     // q (unscaled)
  __bf16* kb   = (__bf16*)take((size_t)NWH * TOKP * HDIM * 2);     // k
  __bf16* vtb  = (__bf16*)take((size_t)NWH * HDIM * TOKP * 2 + 4096);  // v^T + slack
  __bf16* ao   = (__bf16*)take((size_t)MROWS * DIM * 2);           // attn out (windowed)
  float*  hid  = (float*)take((size_t)NTOKF * DIM * 4);            // shortcut + attn
  __bf16* h2   = (__bf16*)take((size_t)NTOKF * DIM * 2);           // LN2
  __bf16* mb   = (__bf16*)take((size_t)NTOKF * MLPD * 2);          // gelu(lin1)
  __bf16* wqkvT  = (__bf16*)take((size_t)(3 * DIM) * DIM * 2);
  __bf16* wprojT = (__bf16*)take((size_t)DIM * DIM * 2);
  __bf16* wl1T   = (__bf16*)take((size_t)MLPD * DIM * 2);
  __bf16* wl2T   = (__bf16*)take((size_t)DIM * MLPD * 2);

  // weights -> bf16, transposed to N-major/K-contiguous
  convT_kernel<<<2048, 256, 0, stream>>>(qkvw, wqkvT, DIM, 3 * DIM);
  convT_kernel<<<1024, 256, 0, stream>>>(projw, wprojT, DIM, DIM);
  convT_kernel<<<2048, 256, 0, stream>>>(l1w, wl1T, DIM, MLPD);
  convT_kernel<<<2048, 256, 0, stream>>>(l2w, wl2T, MLPD, DIM);

  // LN1 + window partition; zero pad rows
  ln_kernel<<<NTOKF, 256, 0, stream>>>(x, n1g, n1b, xw, 1);
  padzero_kernel<<<(NWIN * (TOKP - TOK) * DIM + 255) / 256, 256, 0, stream>>>(xw);

  // QKV GEMM: M=53248, N=2304, K=768
  {
    GArgs g = {};
    g.A = xw; g.Bt = wqkvT; g.K = DIM; g.bias = qkvb;
    g.q = qb; g.k = kb; g.v = vtb;
    gemm_kernel<EPI_QKV><<<dim3((3 * DIM) / 128, MROWS / 128), 256, 0, stream>>>(g);
  }

  // attention per (window, head)
  attn_kernel<<<NWH, 416, 0, stream>>>(qb, kb, vtb, relh, relw, ao);

  // proj GEMM + un-window + residual -> hidden (f32)
  {
    GArgs g = {};
    g.A = ao; g.Bt = wprojT; g.K = DIM; g.bias = projb;
    g.resid = x; g.outf = hid;
    gemm_kernel<EPI_PROJ><<<dim3(DIM / 128, MROWS / 128), 256, 0, stream>>>(g);
  }

  // LN2 -> h2 (flat layout, M=50176 divisible by 128)
  ln_kernel<<<NTOKF, 256, 0, stream>>>(hid, n2g, n2b, h2, 0);

  // lin1 GEMM + GELU -> mb
  {
    GArgs g = {};
    g.A = h2; g.Bt = wl1T; g.K = DIM; g.bias = l1b;
    g.outb = mb;
    gemm_kernel<EPI_L1><<<dim3(MLPD / 128, NTOKF / 128), 256, 0, stream>>>(g);
  }

  // lin2 GEMM + bias + residual -> d_out (f32)
  {
    GArgs g = {};
    g.A = mb; g.Bt = wl2T; g.K = MLPD; g.bias = l2b;
    g.resid = hid; g.outf = out;
    gemm_kernel<EPI_L2><<<dim3(DIM / 128, NTOKF / 128), 256, 0, stream>>>(g);
  }
}
